// COMET_26439818674626
// MI455X (gfx1250) — matrix operand order repaired and checked by reading, not compiled, not run on hardware
//
#include <hip/hip_runtime.h>
#include <hip/hip_bf16.h>
#include <math.h>

// ---------------------------------------------------------------- constants
#define BB 64
#define NN 256
#define TT 96
#define PP 8
#define SS 4
#define DD 128
#define HH 8
#define KK 16
#define PL 24
#define LL 23          // num patches
#define LPAD 32        // padded rows for WMMA (rows 23..31 are exact zeros)

// LDS pitches (in elements) chosen so row stride is a multiple of 16B and
// lane-strided accesses spread across the 64 LDS banks.
#define PA  136   // A (h0 bf16)      [32][128]+8
#define PB1 136   // W1^T bf16        [256][128]+8
#define PG  264   // G (gelu) bf16    [32][256]+8
#define PW2 264   // W2^T bf16        [128][256]+8
#define PH  132   // h0 f32           [32][128]+4

typedef __attribute__((ext_vector_type(16))) __bf16 v16bf;
typedef __attribute__((ext_vector_type(8)))  __bf16 v8bf;
typedef __attribute__((ext_vector_type(8)))  float  v8f;

// obs_mask is a jnp bool array -> 1 byte per element on device
typedef unsigned char mask_t;

// ---------------------------------------------------------------- fast math
// branch-free exp: e^x = 2^(x*log2(e)); args are <= 0 in softmax so no range
// reduction needed (underflow -> 0, matching reference semantics closely).
__device__ __forceinline__ float fast_exp(float x) {
    return __builtin_amdgcn_exp2f(x * 1.4426950408889634f);
}

// CDNA5 has V_TANH_F32 (TRANS op). Fall back to branch-free exp2/rcp form,
// which saturates to +-1 at the extremes with no exec-mask branches.
__device__ __forceinline__ float fast_tanh(float x) {
#if __has_builtin(__builtin_amdgcn_tanhf)
    return __builtin_amdgcn_tanhf(x);
#else
    float e = __builtin_amdgcn_exp2f(x * 2.8853900817779268f);  // 2*log2(e)
    return 1.0f - 2.0f * __builtin_amdgcn_rcpf(e + 1.0f);
#endif
}

__device__ __forceinline__ float gelu_tanh(float x) {
    // jax.nn.gelu default (approximate=True)
    float x3 = x * x * x;
    return 0.5f * x * (1.0f + fast_tanh(0.7978845608028654f * (x + 0.044715f * x3)));
}

// ---------------------------------------------------------------- WMMA frags
// A-matrix fragment (16x32 bf16). ISA layout: lanes 0-15 = rows, elems 0..7 ->
// K=0..7, elems 8..15 -> K=16..23; lanes 16-31 get K offset +8.
__device__ __forceinline__ v16bf load_a_frag(const __bf16* base, int pitch,
                                             int mrow0, int kk, int lane) {
    int row = mrow0 + (lane & 15);
    int kb  = kk + ((lane >> 4) << 3);
    const __bf16* p = base + row * pitch + kb;
    v8bf lo = *(const v8bf*)(p);
    v8bf hi = *(const v8bf*)(p + 16);
    v16bf a;
#pragma unroll
    for (int i = 0; i < 8; ++i) { a[i] = lo[i]; a[i + 8] = hi[i]; }
    return a;
}

// B-matrix fragment (32x16 bf16). Lanes 0-15: elems 0..7 -> K=0..7, elems 8..15 ->
// K=16..23; lanes 16-31 get K offset +8. Weights stored transposed [n][k] in LDS so the
// 16 K-values per lane are contiguous (two 16B loads).
__device__ __forceinline__ v16bf load_b_frag(const __bf16* base, int pitch,
                                             int ncol0, int kk, int lane) {
    int n  = ncol0 + (lane & 15);
    int kb = kk + ((lane >> 4) << 3);
    const __bf16* p = base + n * pitch + kb;
    v8bf lo = *(const v8bf*)(p);
    v8bf hi = *(const v8bf*)(p + 16);
    v16bf b;
#pragma unroll
    for (int i = 0; i < 8; ++i) { b[i] = lo[i]; b[i + 8] = hi[i]; }
    return b;
}

// ---------------------------------------------------------------- kernel 0
// f32 -> bf16 transposed weight staging: wt1t[n][k]=W_t1[k][n], wt2t[n][k]=W_t2[k][n]
__global__ void k_convert_weights(const float* __restrict__ W1,
                                  const float* __restrict__ W2,
                                  __bf16* __restrict__ wt1t,
                                  __bf16* __restrict__ wt2t) {
    int i = blockIdx.x * blockDim.x + threadIdx.x;
    if (i < DD * 2 * DD) {                       // 128*256 : W1^T [256][128]
        int n = i >> 7, k = i & 127;
        wt1t[i] = (__bf16)W1[k * (2 * DD) + n];
    } else if (i < 2 * DD * 2 * DD) {            // W2^T [128][256]
        int j = i - DD * 2 * DD;
        int n = j >> 8, k = j & 255;
        wt2t[j] = (__bf16)W2[k * DD + n];
    }
}

// ---------------------------------------------------------------- kernel 1
// stable partition: observed variate indices first, then unobserved
__global__ void k_mask_scan(const mask_t* __restrict__ obs_mask,
                            int* __restrict__ obs_idx, int* __restrict__ n_obs) {
    int b = blockIdx.x * blockDim.x + threadIdx.x;
    if (b >= BB) return;
    int cnt = 0;
    for (int n = 0; n < NN; ++n)
        if (obs_mask[b * NN + n]) obs_idx[b * NN + cnt++] = n;
    n_obs[b] = cnt;
    int p = cnt;
    for (int n = 0; n < NN; ++n)
        if (!obs_mask[b * NN + n]) obs_idx[b * NN + p++] = n;
}

// ---------------------------------------------------------------- kernel 2
// Fused: patch embed -> h0, h = h0 + gelu(h0@W1)@W2 via bf16 WMMA, mean over L.
// 128 threads (4 waves), loops over 64 (b,n) pairs per block; weights in LDS.
__global__ void k_fused_mlp_mean(const float* __restrict__ x_full,
                                 const mask_t* __restrict__ obs_mask,
                                 const float* __restrict__ W_patch,
                                 const float* __restrict__ b_patch,
                                 const __bf16* __restrict__ wt1t,
                                 const __bf16* __restrict__ wt2t,
                                 float* __restrict__ h_mean) {
    extern __shared__ __align__(16) char smem[];
    __bf16* sW1t = (__bf16*)smem;              // [256][PB1]
    __bf16* sW2t = sW1t + NN * PB1;            // [128][PW2]
    __bf16* sAbf = sW2t + DD * PW2;            // [32][PA]
    __bf16* sGbf = sAbf + LPAD * PA;           // [32][PG]
    float*  sH0  = (float*)(sGbf + LPAD * PG); // [32][PH]
    float*  sX   = sH0 + LPAD * PH;            // [128]
    float*  sWp  = sX + 128;                   // [8][128]
    float*  sBp  = sWp + PP * DD;              // [128]
    float*  sCol = sBp + DD;                   // [128] column sums

    const int tid  = threadIdx.x;   // 0..127
    const int lane = tid & 31;
    const int wave = tid >> 5;

    // stage weights once per block
    for (int i = tid; i < NN * DD; i += 128)
        sW1t[(i >> 7) * PB1 + (i & 127)] = wt1t[i];
    for (int i = tid; i < DD * 2 * DD; i += 128)
        sW2t[(i >> 8) * PW2 + (i & 255)] = wt2t[i];
    for (int i = tid; i < PP * DD; i += 128) sWp[i] = W_patch[i];
    sBp[tid] = b_patch[tid];

    const int pair0 = blockIdx.x * 64;
    for (int pp = 0; pp < 64; ++pp) {
        const int pair = pair0 + pp;        // = b*NN + n
        __syncthreads();
        if (tid < TT) {
            float xv = x_full[pair * TT + tid];
            sX[tid] = obs_mask[pair] ? xv : 0.0f;
        }
        sCol[tid] = 0.0f;
        // prefetch next pair's time series into cache (global_prefetch_b8)
        if (pp + 1 < 64 && tid == 0)
            __builtin_prefetch(&x_full[(pair + 1) * TT], 0, 3);
        __syncthreads();

        // ---- patch embedding: h0[l][d] (rows >= LL are exact zeros)
#pragma unroll
        for (int e = 0; e < 32; ++e) {
            int idx = tid + e * 128;
            int l = idx >> 7, d = idx & 127;
            float acc = 0.0f;
            if (l < LL) {
                acc = sBp[d];
#pragma unroll
                for (int p = 0; p < PP; ++p)
                    acc += sX[l * SS + p] * sWp[p * DD + d];
            }
            sH0[l * PH + d]  = acc;
            sAbf[l * PA + d] = (__bf16)acc;
        }
        __syncthreads();

        // ---- matmul1 [32,128]@[128,256] + gelu, bf16 WMMA
        for (int t = wave; t < 32; t += 4) {     // 2 M-tiles x 16 N-tiles
            int mt = t >> 4, nt = t & 15;
            v8f acc = {};
#pragma unroll
            for (int kk = 0; kk < DD; kk += 32) {
                v16bf a = load_a_frag(sAbf, PA, mt * 16, kk, lane);
                v16bf b = load_b_frag(sW1t, PB1, nt * 16, kk, lane);
                acc = __builtin_amdgcn_wmma_f32_16x16x32_bf16(
                    false, a, false, b, (short)0, acc, false, false);
            }
            int col   = nt * 16 + (lane & 15);
            int rbase = mt * 16 + ((lane >> 4) << 3);
#pragma unroll
            for (int r = 0; r < 8; ++r)
                sGbf[(rbase + r) * PG + col] = (__bf16)gelu_tanh(acc[r]);
        }
        __syncthreads();

        // ---- matmul2 [32,256]@[256,128] + residual, fold mean over rows
        for (int t = wave; t < 16; t += 4) {     // 2 M-tiles x 8 N-tiles
            int mt = t >> 3, nt = t & 7;
            v8f acc = {};
#pragma unroll
            for (int kk = 0; kk < 2 * DD; kk += 32) {
                v16bf a = load_a_frag(sGbf, PG, mt * 16, kk, lane);
                v16bf b = load_b_frag(sW2t, PW2, nt * 16, kk, lane);
                acc = __builtin_amdgcn_wmma_f32_16x16x32_bf16(
                    false, a, false, b, (short)0, acc, false, false);
            }
            int col   = nt * 16 + (lane & 15);
            int rbase = mt * 16 + ((lane >> 4) << 3);
            float psum = 0.0f;
#pragma unroll
            for (int r = 0; r < 8; ++r)
                psum += sH0[(rbase + r) * PH + col] + acc[r];
            atomicAdd(&sCol[col], psum);         // ds_add_f32
        }
        __syncthreads();
        h_mean[pair * DD + tid] = sCol[tid] * (1.0f / (float)LL);
    }
}

// ---------------------------------------------------------------- kernel 3
// tokens[b,i,:] = h_mean[b, obs_idx[b,i], :] + var_embed[obs_idx[b,i], :]
__global__ void k_tokens(const float* __restrict__ h_mean,
                         const float* __restrict__ var_embed,
                         const int* __restrict__ obs_idx,
                         float* __restrict__ tokens) {
    int i = blockIdx.x * blockDim.x + threadIdx.x;
    if (i >= BB * NN * DD) return;
    int d = i & (DD - 1);
    int bn = i >> 7;
    int b = bn >> 8;
    int src = obs_idx[bn];
    tokens[i] = h_mean[(b * NN + src) * DD + d] + var_embed[src * DD + d];
}

// ---------------------------------------------------------------- kernel 4
// one block per (b, head); thread = query row; two-pass masked softmax
__global__ void k_attention(const float* __restrict__ tokens,
                            const float* __restrict__ Wq,
                            const float* __restrict__ Wk,
                            const float* __restrict__ Wv,
                            const int* __restrict__ n_obs,
                            float* __restrict__ attn_out) {
    int b = blockIdx.x >> 3;
    int h = blockIdx.x & 7;
    int t = threadIdx.x;              // 0..255
    __shared__ float sK[NN * 16], sV[NN * 16];
    __shared__ float sWq[DD * 16], sWk[DD * 16], sWv[DD * 16];
    for (int i = t; i < DD * 16; i += NN) {
        int d = i >> 4, c = i & 15;
        sWq[i] = Wq[d * DD + h * 16 + c];
        sWk[i] = Wk[d * DD + h * 16 + c];
        sWv[i] = Wv[d * DD + h * 16 + c];
    }
    __syncthreads();

    float q[16], kreg[16], vreg[16];
#pragma unroll
    for (int c = 0; c < 16; ++c) { q[c] = 0.f; kreg[c] = 0.f; vreg[c] = 0.f; }
    const float* tok = tokens + (b * NN + t) * DD;
    for (int d = 0; d < DD; ++d) {
        float x = tok[d];
#pragma unroll
        for (int c = 0; c < 16; ++c) {
            q[c]    += x * sWq[d * 16 + c];
            kreg[c] += x * sWk[d * 16 + c];
            vreg[c] += x * sWv[d * 16 + c];
        }
    }
#pragma unroll
    for (int c = 0; c < 16; ++c) { sK[t * 16 + c] = kreg[c]; sV[t * 16 + c] = vreg[c]; }
    __syncthreads();

    const int nv = n_obs[b];
    float m = -1e30f;
    for (int j = 0; j < NN; ++j) {
        float s = 0.f;
#pragma unroll
        for (int c = 0; c < 16; ++c) s += q[c] * sK[j * 16 + c];
        s *= 0.25f;                               // dh^-0.5
        if (j >= nv) s = -1e9f;
        m = fmaxf(m, s);
    }
    float sum = 0.f, o[16];
#pragma unroll
    for (int c = 0; c < 16; ++c) o[c] = 0.f;
    for (int j = 0; j < NN; ++j) {
        float s = 0.f;
#pragma unroll
        for (int c = 0; c < 16; ++c) s += q[c] * sK[j * 16 + c];
        s *= 0.25f;
        if (j >= nv) s = -1e9f;
        float e = fast_exp(s - m);                // single v_exp_f32
        sum += e;
#pragma unroll
        for (int c = 0; c < 16; ++c) o[c] += e * sV[j * 16 + c];
    }
    float inv = __builtin_amdgcn_rcpf(sum);
#pragma unroll
    for (int c = 0; c < 16; ++c)
        attn_out[(b * NN + t) * DD + h * 16 + c] = o[c] * inv;
}

// ---------------------------------------------------------------- kernel 5
// out = attn_out@Wo + tokens ; Q_sub = mean over valid rows. One block per b.
__global__ void k_proj_qsub(const float* __restrict__ attn_out,
                            const float* __restrict__ tokens,
                            const float* __restrict__ Wo,
                            const int* __restrict__ n_obs,
                            float* __restrict__ Qsub) {
    int b = blockIdx.x;
    int t = threadIdx.x;              // 0..255 (row)
    __shared__ float sWo[DD * 32];
    __shared__ float sQ[DD];
    if (t < DD) sQ[t] = 0.0f;
    const int nv = n_obs[b];
    const float* arow = attn_out + (b * NN + t) * DD;
    const float* trow = tokens + (b * NN + t) * DD;
    for (int dc = 0; dc < DD; dc += 32) {
        __syncthreads();
        for (int i = t; i < DD * 32; i += NN) {
            int k = i >> 5, d = i & 31;
            sWo[i] = Wo[k * DD + dc + d];
        }
        __syncthreads();
        float acc[32];
#pragma unroll
        for (int d = 0; d < 32; ++d) acc[d] = trow[dc + d];
        for (int k = 0; k < DD; ++k) {
            float a = arow[k];
#pragma unroll
            for (int d = 0; d < 32; ++d) acc[d] += a * sWo[k * 32 + d];
        }
        if (t < nv) {
#pragma unroll
            for (int d = 0; d < 32; ++d) atomicAdd(&sQ[dc + d], acc[d]);
        }
    }
    __syncthreads();
    if (t < DD) Qsub[b * DD + t] = sQ[t] / (float)nv;
}

// ---------------------------------------------------------------- kernel 6
// codebook: w_sub = softmax(Qsub@C^T / tau); z_ctx = w@C; fill = z_ctx@W_dec
__global__ void k_codebook(const float* __restrict__ Qsub,
                           const float* __restrict__ C,
                           const float* __restrict__ W_dec,
                           const int* __restrict__ n_obs,
                           float* __restrict__ w_sub, float* __restrict__ conf,
                           float* __restrict__ fill, float* __restrict__ alpha) {
    int b = threadIdx.x;
    if (b >= BB) return;
    const float* q = Qsub + b * DD;
    float w[KK];
    float m = -1e30f;
    for (int k = 0; k < KK; ++k) {
        float s = 0.f;
        for (int d = 0; d < DD; ++d) s += q[d] * C[k * DD + d];
        w[k] = s * 2.0f;                 // / TAU (0.5)
        m = fmaxf(m, w[k]);
    }
    float sum = 0.f;
    for (int k = 0; k < KK; ++k) { w[k] = fast_exp(w[k] - m); sum += w[k]; }
    float inv = __builtin_amdgcn_rcpf(sum);
    float wmax = 0.f;
    for (int k = 0; k < KK; ++k) {
        w[k] *= inv;
        wmax = fmaxf(wmax, w[k]);
        w_sub[b * KK + k] = w[k];
    }
    float cv = wmax * ((float)n_obs[b] / (float)NN);
    conf[b]  = cv;
    alpha[b] = 0.1f + 0.9f * cv;
    float z[DD];
    for (int d = 0; d < DD; ++d) {
        float s = 0.f;
        for (int k = 0; k < KK; ++k) s += w[k] * C[k * DD + d];
        z[d] = s;
    }
    for (int dd = 0; dd < DD; ++dd) {
        float s = 0.f;
        for (int d = 0; d < DD; ++d) s += z[d] * W_dec[d * DD + dd];
        fill[b * DD + dd] = s;
    }
}

// ---------------------------------------------------------------- kernel 7
// y_hat = (h_mean + alpha*(1-obs)*fill) @ W_head
__global__ void k_head(const float* __restrict__ h_mean,
                       const mask_t* __restrict__ obs_mask,
                       const float* __restrict__ fill,
                       const float* __restrict__ alpha,
                       const float* __restrict__ W_head,
                       float* __restrict__ y) {
    int i = blockIdx.x * blockDim.x + threadIdx.x;
    if (i >= BB * NN * PL) return;
    int pl = i % PL;
    int bn = i / PL;
    int b  = bn >> 8;
    float a = obs_mask[bn] ? 0.0f : alpha[b];
    const float* hr = h_mean + bn * DD;
    const float* fr = fill + b * DD;
    float acc = 0.f;
    for (int d = 0; d < DD; ++d)
        acc += (hr[d] + a * fr[d]) * W_head[d * PL + pl];
    y[i] = acc;
}

// ---------------------------------------------------------------- launcher
extern "C" void kernel_launch(void* const* d_in, const int* in_sizes, int n_in,
                              void* d_out, int out_size, void* d_ws, size_t ws_size,
                              hipStream_t stream) {
    const float*  x_full    = (const float*)d_in[0];
    const mask_t* obs_mask  = (const mask_t*)d_in[1];   // jnp bool -> 1 byte
    const float*  W_patch   = (const float*)d_in[2];
    const float*  b_patch   = (const float*)d_in[3];
    const float*  W_t1      = (const float*)d_in[4];
    const float*  W_t2      = (const float*)d_in[5];
    const float*  var_embed = (const float*)d_in[6];
    const float*  Wq        = (const float*)d_in[7];
    const float*  Wk        = (const float*)d_in[8];
    const float*  Wv        = (const float*)d_in[9];
    const float*  Wo        = (const float*)d_in[10];
    const float*  C         = (const float*)d_in[11];
    const float*  W_dec     = (const float*)d_in[12];
    const float*  W_head    = (const float*)d_in[13];

    char* ws = (char*)d_ws;
    __bf16* wt1t  = (__bf16*)(ws);                       //  64 KB
    __bf16* wt2t  = (__bf16*)(ws + 65536);               //  64 KB
    float* h_mean = (float*)(ws + 131072);               //   8 MB
    float* tokens = (float*)(ws + 131072 + 8388608);     //   8 MB
    float* attn_o = (float*)(ws + 131072 + 16777216);    //   8 MB
    int*   obs_idx= (int*)  (ws + 131072 + 25165824);    //  64 KB
    int*   n_obs  = (int*)  ((char*)obs_idx + 65536);    // 256 B
    float* fill   = (float*)((char*)n_obs + 256);        //  32 KB
    float* alpha  = (float*)((char*)fill + 32768);       // 256 B

    float* y_hat = (float*)d_out;
    float* Qsub  = y_hat + BB * NN * PL;
    float* w_sub = Qsub + BB * DD;
    float* conf  = w_sub + BB * KK;

    k_convert_weights<<<(2 * DD * 2 * DD + 255) / 256, 256, 0, stream>>>(
        W_t1, W_t2, wt1t, wt2t);
    k_mask_scan<<<1, 64, 0, stream>>>(obs_mask, obs_idx, n_obs);

    // dynamic LDS: bf16 region 162816 B + f32 region 22528 B
    size_t smem = (size_t)(NN * PB1 + DD * PW2 + LPAD * PA + LPAD * PG) * 2
                + (size_t)(LPAD * PH + 128 + PP * DD + DD + DD) * 4;
    k_fused_mlp_mean<<<256, 128, smem, stream>>>(
        x_full, obs_mask, W_patch, b_patch, wt1t, wt2t, h_mean);

    k_tokens<<<(BB * NN * DD + 255) / 256, 256, 0, stream>>>(
        h_mean, var_embed, obs_idx, tokens);
    k_attention<<<BB * HH, NN, 0, stream>>>(tokens, Wq, Wk, Wv, n_obs, attn_o);
    k_proj_qsub<<<BB, NN, 0, stream>>>(attn_o, tokens, Wo, n_obs, Qsub);
    k_codebook<<<1, 64, 0, stream>>>(Qsub, C, W_dec, n_obs, w_sub, conf, fill, alpha);
    k_head<<<(BB * NN * PL + 255) / 256, 256, 0, stream>>>(
        h_mean, obs_mask, fill, alpha, W_head, y_hat);
}
